// NMT_47193100648743
// MI455X (gfx1250) — matrix operand order repaired and checked by reading, not compiled, not run on hardware
//
#include <hip/hip_runtime.h>
#include <cstdint>
#include <cstddef>

#define S_ 50
#define T_ 50
#define B_ 64
#define E_ 300
#define EP_ 320          // E padded to multiple of 32
#define EH_ 512
#define DH_ 1024
#define V_ 23262
#define VP_ 23264        // V padded to multiple of 16

typedef __bf16 bf16;
typedef bf16 v16bf __attribute__((ext_vector_type(16)));
typedef bf16 v8bf  __attribute__((ext_vector_type(8)));
typedef float v8f  __attribute__((ext_vector_type(8)));

__device__ __forceinline__ float sigmoidf_(float x) { return 1.0f / (1.0f + __expf(-x)); }

// ---------------------------------------------------------------------------
// WMMA GEMM: C[M x Nreal] = A[M x K] * W[N x K]^T  (+ bias[n]) (+ C if beta)
// A row-major (lda = padded K, mult of 32), W row-major (ldw = padded K).
// One wave computes a 64x16 strip of C (4 M-tiles sharing one B fragment per
// K-chunk) via v_wmma_f32_16x16x32_bf16. M must be a multiple of 64.
// ---------------------------------------------------------------------------
__global__ void k_gemm(const bf16* __restrict__ A, int lda,
                       const bf16* __restrict__ W, int ldw,
                       float* __restrict__ C, int ldc,
                       const float* __restrict__ bias,
                       int nTiles, int Nreal, int K, int beta)
{
    int lane  = threadIdx.x & 31;
    int wave  = threadIdx.x >> 5;
    int nTile = blockIdx.x * (blockDim.x >> 5) + wave;
    if (nTile >= nTiles) return;           // uniform per wave: EXEC stays all-ones
    int mBase = blockIdx.y * 64;
    int hi  = lane >> 4;                   // 0: lanes 0-15, 1: lanes 16-31
    int l15 = lane & 15;

    // A fragment: lane holds row M; lo-half lanes carry K 0..7 & 16..23,
    // hi-half lanes carry K 8..15 & 24..31 (ISA 16-bit A 16x32 layout).
    const bf16* ap0 = A + (size_t)(mBase + l15) * lda + hi * 8;
    // B fragment: lane holds column N; lo-half lanes carry K 0..7 & 16..23,
    // hi-half lanes carry K 8..15 & 24..31 (same K order as A).
    const bf16* wp = W + (size_t)(nTile * 16 + l15) * ldw + hi * 8;
    size_t mstep = (size_t)16 * lda;

    v8f acc0 = {}, acc1 = {}, acc2 = {}, acc3 = {};
    for (int kc = 0; kc < K; kc += 32) {
        v8bf b0 = *(const v8bf*)(wp + kc);
        v8bf b1 = *(const v8bf*)(wp + kc + 16);
        v16bf b;
#pragma unroll
        for (int i = 0; i < 8; ++i) { b[i] = b0[i]; b[8 + i] = b1[i]; }

        v8bf a0, a1;
        v16bf a;
        a0 = *(const v8bf*)(ap0 + kc);
        a1 = *(const v8bf*)(ap0 + kc + 16);
#pragma unroll
        for (int i = 0; i < 8; ++i) { a[i] = a0[i]; a[8 + i] = a1[i]; }
        acc0 = __builtin_amdgcn_wmma_f32_16x16x32_bf16(false, a, false, b, (short)0, acc0, false, false);

        a0 = *(const v8bf*)(ap0 + mstep + kc);
        a1 = *(const v8bf*)(ap0 + mstep + kc + 16);
#pragma unroll
        for (int i = 0; i < 8; ++i) { a[i] = a0[i]; a[8 + i] = a1[i]; }
        acc1 = __builtin_amdgcn_wmma_f32_16x16x32_bf16(false, a, false, b, (short)0, acc1, false, false);

        a0 = *(const v8bf*)(ap0 + 2 * mstep + kc);
        a1 = *(const v8bf*)(ap0 + 2 * mstep + kc + 16);
#pragma unroll
        for (int i = 0; i < 8; ++i) { a[i] = a0[i]; a[8 + i] = a1[i]; }
        acc2 = __builtin_amdgcn_wmma_f32_16x16x32_bf16(false, a, false, b, (short)0, acc2, false, false);

        a0 = *(const v8bf*)(ap0 + 3 * mstep + kc);
        a1 = *(const v8bf*)(ap0 + 3 * mstep + kc + 16);
#pragma unroll
        for (int i = 0; i < 8; ++i) { a[i] = a0[i]; a[8 + i] = a1[i]; }
        acc3 = __builtin_amdgcn_wmma_f32_16x16x32_bf16(false, a, false, b, (short)0, acc3, false, false);
    }

    int n = nTile * 16 + l15;
    if (n >= Nreal) return;
    float bv = bias ? bias[n] : 0.0f;
    v8f accs[4] = {acc0, acc1, acc2, acc3};
#pragma unroll
    for (int mt = 0; mt < 4; ++mt) {
        int m0 = mBase + mt * 16 + hi * 8;  // C/D layout: vgpr j -> row m0+j
#pragma unroll
        for (int j = 0; j < 8; ++j) {
            size_t off = (size_t)(m0 + j) * ldc + n;
            float v = accs[mt][j] + bv;
            if (beta) v += C[off];
            C[off] = v;
        }
    }
}

// ---------------------------------------------------------------------------
// Weight convert: f32 (N x K slice, row stride ld, col offset coff) ->
// zero-padded bf16 (Np x Kp)
// ---------------------------------------------------------------------------
__global__ void k_convw(const float* __restrict__ s, bf16* __restrict__ d,
                        int N, int K, int Kp, int ld, int coff, int total)
{
    int idx = blockIdx.x * blockDim.x + threadIdx.x;
    if (idx >= total) return;
    int r = idx / Kp, k = idx % Kp;
    float v = (r < N && k < K) ? s[(size_t)r * ld + coff + k] : 0.0f;
    d[idx] = (bf16)v;
}

__global__ void k_zero_f32(float* p, int n) { int i = blockIdx.x * blockDim.x + threadIdx.x; if (i < n) p[i] = 0.0f; }
__global__ void k_zero_bf16(bf16* p, int n) { int i = blockIdx.x * blockDim.x + threadIdx.x; if (i < n) p[i] = (bf16)0.0f; }

// Encoder embedding gather -> padded bf16, plus time-reversed copy for bwd dir
__global__ void k_embed_src(const int* __restrict__ src, const float* __restrict__ emb,
                            bf16* __restrict__ x, bf16* __restrict__ xr)
{
    int row = blockIdx.x;                 // row = s*B + b
    int s = row / B_, b = row % B_;
    int k = threadIdx.x;                  // 0..319
    float v = (k < E_) ? emb[(size_t)src[row] * E_ + k] : 0.0f;
    bf16 bv = (bf16)v;
    x[(size_t)row * EP_ + k] = bv;
    xr[((size_t)(S_ - 1 - s) * B_ + b) * EP_ + k] = bv;
}

// Decoder embedding gather (first T-1 target steps) -> padded bf16
__global__ void k_embed_trg(const int* __restrict__ trg, const float* __restrict__ emb,
                            bf16* __restrict__ e)
{
    int row = blockIdx.x;                 // row = t*B + b, t < T-1
    int k = threadIdx.x;
    float v = (k < E_) ? emb[(size_t)trg[row] * E_ + k] : 0.0f;
    e[(size_t)row * EP_ + k] = (bf16)v;
}

// LSTM elementwise: gates (B x 4H) [+ xpart], c' = sig(f)*c + sig(i)*tanh(g),
// h = sig(o)*tanh(c'). Writes f32 h,c, bf16 h, optional hs slot (stride 1024).
__global__ void k_lstm(const float* __restrict__ gates, const float* __restrict__ xpart,
                       const float* __restrict__ cprev,
                       float* __restrict__ hout, float* __restrict__ cout,
                       bf16* __restrict__ hbf, float* __restrict__ hsdst, int H)
{
    int idx = blockIdx.x * blockDim.x + threadIdx.x;
    if (idx >= B_ * H) return;
    int b = idx / H, j = idx % H;
    size_t g0 = (size_t)b * 4 * H + j;
    float gi = gates[g0], gf = gates[g0 + H], gg = gates[g0 + 2 * H], go = gates[g0 + 3 * H];
    if (xpart) { gi += xpart[g0]; gf += xpart[g0 + H]; gg += xpart[g0 + 2 * H]; go += xpart[g0 + 3 * H]; }
    float c = sigmoidf_(gf) * cprev[idx] + sigmoidf_(gi) * tanhf(gg);
    float h = sigmoidf_(go) * tanhf(c);
    cout[idx] = c; hout[idx] = h; hbf[idx] = (bf16)h;
    if (hsdst) hsdst[(size_t)b * 1024 + j] = h;
}

// Decoder init: h = reshape(stack([hf,hb],0),(B,DH)) (exact reference reshape)
__global__ void k_decinit(const float* hf, const float* cf, const float* hb, const float* cb,
                          float* h0, float* c0, bf16* h0b)
{
    int idx = blockIdx.x * blockDim.x + threadIdx.x;
    if (idx >= B_ * DH_) return;
    int dir = idx >> 15;                  // flat / (B*EH) with B*EH = 32768
    int rem = idx & 32767;
    int bb = rem >> 9, dd = rem & 511;
    float hv = dir ? hb[bb * 512 + dd] : hf[bb * 512 + dd];
    float cv = dir ? cb[bb * 512 + dd] : cf[bb * 512 + dd];
    h0[idx] = hv; c0[idx] = cv; h0b[idx] = (bf16)hv;
}

// scores[s,b] = dot(ouths[s,b,:], hWi[b,:]) over 1024
__global__ void k_scores(const float* __restrict__ ouths, const float* __restrict__ hWi,
                         float* __restrict__ sc)
{
    __shared__ float sd[128];
    int r = blockIdx.x;                   // r = s*B + b
    int b = r & (B_ - 1);
    const float* x = ouths + (size_t)r * 1024;
    const float* y = hWi + (size_t)b * 1024;
    float acc = 0.0f;
    for (int k = threadIdx.x; k < 1024; k += 128) acc += x[k] * y[k];
    sd[threadIdx.x] = acc; __syncthreads();
    for (int s = 64; s > 0; s >>= 1) { if (threadIdx.x < s) sd[threadIdx.x] += sd[threadIdx.x + s]; __syncthreads(); }
    if (threadIdx.x == 0) sc[r] = sd[0];
}

// softmax over batch axis (axis=1 in reference) for each source position s
__global__ void k_softmaxb(const float* __restrict__ sc, float* __restrict__ a)
{
    __shared__ float sd[64];
    int s = blockIdx.x, b = threadIdx.x;
    float v = sc[s * B_ + b];
    sd[b] = v; __syncthreads();
    for (int k = 32; k > 0; k >>= 1) { if (b < k) sd[b] = fmaxf(sd[b], sd[b + k]); __syncthreads(); }
    float m = sd[0]; __syncthreads();
    float e = __expf(v - m);
    sd[b] = e; __syncthreads();
    for (int k = 32; k > 0; k >>= 1) { if (b < k) sd[b] += sd[b + k]; __syncthreads(); }
    a[s * B_ + b] = e / sd[0];
}

// s_t[b,d] = sum_s a[s,b] * ouths[s,b,d]
__global__ void k_st(const float* __restrict__ a, const float* __restrict__ ouths,
                     float* __restrict__ st)
{
    int idx = blockIdx.x * blockDim.x + threadIdx.x;
    if (idx >= B_ * 1024) return;
    int b = idx >> 10, d = idx & 1023;
    float acc = 0.0f;
    for (int s = 0; s < S_; ++s) acc += a[s * B_ + b] * ouths[((size_t)(s * B_ + b)) * 1024 + d];
    st[idx] = acc;
}

// cat = bf16([s_t, h])  (B x 2048)
__global__ void k_concat(const float* __restrict__ st, const float* __restrict__ h,
                         bf16* __restrict__ cat)
{
    int idx = blockIdx.x * blockDim.x + threadIdx.x;
    if (idx >= B_ * 2048) return;
    int b = idx >> 11, k = idx & 2047;
    float v = (k < 1024) ? st[b * 1024 + k] : h[b * 1024 + (k - 1024)];
    cat[idx] = (bf16)v;
}

__global__ void k_tanhbf(const float* __restrict__ s, bf16* __restrict__ d, int n)
{
    int i = blockIdx.x * blockDim.x + threadIdx.x;
    if (i < n) d[i] = (bf16)tanhf(s[i]);
}

// in-place log_softmax over one vocab row per block
__global__ void k_logsoftmax(float* __restrict__ base)
{
    __shared__ float sd[256];
    float* row = base + (size_t)blockIdx.x * V_;
    int t = threadIdx.x;
    float m = -3.0e38f;
    for (int k = t; k < V_; k += 256) m = fmaxf(m, row[k]);
    sd[t] = m; __syncthreads();
    for (int s = 128; s > 0; s >>= 1) { if (t < s) sd[t] = fmaxf(sd[t], sd[t + s]); __syncthreads(); }
    m = sd[0]; __syncthreads();
    float acc = 0.0f;
    for (int k = t; k < V_; k += 256) acc += __expf(row[k] - m);
    sd[t] = acc; __syncthreads();
    for (int s = 128; s > 0; s >>= 1) { if (t < s) sd[t] += sd[t + s]; __syncthreads(); }
    float lse = m + __logf(sd[0]);
    for (int k = t; k < V_; k += 256) row[k] = row[k] - lse;
}

// ---------------------------------------------------------------------------
extern "C" void kernel_launch(void* const* d_in, const int* in_sizes, int n_in,
                              void* d_out, int out_size, void* d_ws, size_t ws_size,
                              hipStream_t stream)
{
    (void)in_sizes; (void)n_in; (void)out_size; (void)ws_size;
    const int*   src    = (const int*)d_in[0];
    const int*   trg    = (const int*)d_in[1];
    const float* emb_en = (const float*)d_in[2];
    const float* emb_de = (const float*)d_in[3];
    const float* Wihf   = (const float*)d_in[4];
    const float* Whhf   = (const float*)d_in[5];
    const float* b_f    = (const float*)d_in[6];
    const float* Wihb   = (const float*)d_in[7];
    const float* Whhb   = (const float*)d_in[8];
    const float* b_b    = (const float*)d_in[9];
    const float* Wihd   = (const float*)d_in[10];
    const float* Whhd   = (const float*)d_in[11];
    const float* b_d    = (const float*)d_in[12];
    const float* Wgen   = (const float*)d_in[13];
    const float* bgen   = (const float*)d_in[14];
    const float* Wi     = (const float*)d_in[15];
    const float* Wo     = (const float*)d_in[16];
    float* out = (float*)d_out;

    char* p = (char*)d_ws;
    auto alloc = [&](size_t bytes) { char* r = p; p += (bytes + 255) & ~(size_t)255; return r; };

    bf16*  x_bf   = (bf16*)alloc((size_t)S_ * B_ * EP_ * 2);
    bf16*  xr_bf  = (bf16*)alloc((size_t)S_ * B_ * EP_ * 2);
    bf16*  e_bf   = (bf16*)alloc((size_t)(T_ - 1) * B_ * EP_ * 2);
    bf16*  wihf_b = (bf16*)alloc((size_t)2048 * EP_ * 2);
    bf16*  whhf_b = (bf16*)alloc((size_t)2048 * 512 * 2);
    bf16*  wihb_b = (bf16*)alloc((size_t)2048 * EP_ * 2);
    bf16*  whhb_b = (bf16*)alloc((size_t)2048 * 512 * 2);
    bf16*  wihc_b = (bf16*)alloc((size_t)4096 * 1024 * 2);
    bf16*  wihe_b = (bf16*)alloc((size_t)4096 * EP_ * 2);
    bf16*  whhd_b = (bf16*)alloc((size_t)4096 * 1024 * 2);
    bf16*  wgen_b = (bf16*)alloc((size_t)VP_ * 1024 * 2);
    bf16*  wi_b   = (bf16*)alloc((size_t)1024 * 1024 * 2);
    bf16*  wo_b   = (bf16*)alloc((size_t)1024 * 2048 * 2);
    float* Xf     = (float*)alloc((size_t)S_ * B_ * 2048 * 4);
    float* Xb     = (float*)alloc((size_t)S_ * B_ * 2048 * 4);
    float* ouths  = (float*)alloc((size_t)S_ * B_ * 1024 * 4);
    float* hF  = (float*)alloc(B_ * 512 * 4);
    float* cF  = (float*)alloc(B_ * 512 * 4);
    float* hB  = (float*)alloc(B_ * 512 * 4);
    float* cB  = (float*)alloc(B_ * 512 * 4);
    bf16*  hFb = (bf16*)alloc(B_ * 512 * 2);
    bf16*  hBb = (bf16*)alloc(B_ * 512 * 2);
    float* gF  = (float*)alloc(B_ * 2048 * 4);
    float* gB  = (float*)alloc(B_ * 2048 * 4);
    float* hD  = (float*)alloc(B_ * 1024 * 4);
    float* cD  = (float*)alloc(B_ * 1024 * 4);
    bf16*  hDb = (bf16*)alloc(B_ * 1024 * 2);
    float* hWi = (float*)alloc(B_ * 1024 * 4);
    float* sc  = (float*)alloc(S_ * B_ * 4);
    float* aa  = (float*)alloc(S_ * B_ * 4);
    float* st  = (float*)alloc(B_ * 1024 * 4);
    bf16*  catb= (bf16*)alloc(B_ * 2048 * 2);
    float* ctr = (float*)alloc(B_ * 1024 * 4);
    bf16*  ct_b= (bf16*)alloc(B_ * 1024 * 2);
    float* gD  = (float*)alloc(B_ * 4096 * 4);

    auto convw = [&](const float* s, bf16* d, int N, int K, int Kp, int ld, int coff, int Np) {
        int tot = Np * Kp;
        k_convw<<<dim3((tot + 255) / 256), dim3(256), 0, stream>>>(s, d, N, K, Kp, ld, coff, tot);
    };
    auto gemm = [&](const bf16* A, int lda, const bf16* W, int ldw, float* C, int ldc,
                    const float* bias, int M, int Nreal, int Np, int K, int beta) {
        dim3 grid((Np / 16 + 3) / 4, M / 64);           // one wave = 64x16 strip
        k_gemm<<<grid, dim3(128), 0, stream>>>(A, lda, W, ldw, C, ldc, bias, Np / 16, Nreal, K, beta);
    };

    // ---- weight conversion to L2-resident bf16 (W_gen: 47.6 MB, fits L2) ----
    convw(Wihf, wihf_b, 2048, E_, EP_, E_, 0, 2048);
    convw(Whhf, whhf_b, 2048, 512, 512, 512, 0, 2048);
    convw(Wihb, wihb_b, 2048, E_, EP_, E_, 0, 2048);
    convw(Whhb, whhb_b, 2048, 512, 512, 512, 0, 2048);
    convw(Wihd, wihc_b, 4096, 1024, 1024, 1024 + E_, 0, 4096);      // c_t columns
    convw(Wihd, wihe_b, 4096, E_, EP_, 1024 + E_, 1024, 4096);      // embedding columns
    convw(Whhd, whhd_b, 4096, 1024, 1024, 1024, 0, 4096);
    convw(Wgen, wgen_b, V_, 1024, 1024, 1024, 0, VP_);
    convw(Wi,   wi_b,   1024, 1024, 1024, 1024, 0, 1024);
    convw(Wo,   wo_b,   1024, 2048, 2048, 2048, 0, 1024);

    // ---- embeddings ----
    k_embed_src<<<dim3(S_ * B_), dim3(EP_), 0, stream>>>(src, emb_en, x_bf, xr_bf);
    k_embed_trg<<<dim3((T_ - 1) * B_), dim3(EP_), 0, stream>>>(trg, emb_de, e_bf);

    // ---- init state, zero out[0] ----
    k_zero_f32<<<dim3((B_ * 512 + 255) / 256), dim3(256), 0, stream>>>(cF, B_ * 512);
    k_zero_f32<<<dim3((B_ * 512 + 255) / 256), dim3(256), 0, stream>>>(cB, B_ * 512);
    k_zero_bf16<<<dim3((B_ * 512 + 255) / 256), dim3(256), 0, stream>>>(hFb, B_ * 512);
    k_zero_bf16<<<dim3((B_ * 512 + 255) / 256), dim3(256), 0, stream>>>(hBb, B_ * 512);
    k_zero_f32<<<dim3((B_ * V_ + 255) / 256), dim3(256), 0, stream>>>(out, B_ * V_);

    // ---- hoisted encoder input projections: (3200 x 2048, K=320) ----
    gemm(x_bf,  EP_, wihf_b, EP_, Xf, 2048, nullptr, S_ * B_, 2048, 2048, EP_, 0);
    gemm(xr_bf, EP_, wihb_b, EP_, Xb, 2048, nullptr, S_ * B_, 2048, 2048, EP_, 0);

    // ---- encoder recurrence (fwd + bwd) ----
    int lstmEncGrid = (B_ * 512 + 255) / 256;
    for (int t = 0; t < S_; ++t) {
        gemm(hFb, 512, whhf_b, 512, gF, 2048, b_f, B_, 2048, 2048, 512, 0);
        k_lstm<<<dim3(lstmEncGrid), dim3(256), 0, stream>>>(
            gF, Xf + (size_t)t * B_ * 2048, cF, hF, cF, hFb,
            ouths + (size_t)t * B_ * 1024, 512);
        gemm(hBb, 512, whhb_b, 512, gB, 2048, b_b, B_, 2048, 2048, 512, 0);
        k_lstm<<<dim3(lstmEncGrid), dim3(256), 0, stream>>>(
            gB, Xb + (size_t)t * B_ * 2048, cB, hB, cB, hBb,
            ouths + (size_t)(S_ - 1 - t) * B_ * 1024 + 512, 512);
    }

    k_decinit<<<dim3((B_ * DH_ + 255) / 256), dim3(256), 0, stream>>>(hF, cF, hB, cB, hD, cD, hDb);

    // ---- decoder recurrence with attention ----
    int g1k = (B_ * 1024 + 255) / 256;
    int g2k = (B_ * 2048 + 255) / 256;
    for (int t = 0; t < T_ - 1; ++t) {
        gemm(hDb, 1024, wi_b, 1024, hWi, 1024, nullptr, B_, 1024, 1024, 1024, 0);
        k_scores<<<dim3(S_ * B_), dim3(128), 0, stream>>>(ouths, hWi, sc);
        k_softmaxb<<<dim3(S_), dim3(64), 0, stream>>>(sc, aa);
        k_st<<<dim3(g1k), dim3(256), 0, stream>>>(aa, ouths, st);
        k_concat<<<dim3(g2k), dim3(256), 0, stream>>>(st, hD, catb);
        gemm(catb, 2048, wo_b, 2048, ctr, 1024, nullptr, B_, 1024, 1024, 2048, 0);
        k_tanhbf<<<dim3(g1k), dim3(256), 0, stream>>>(ctr, ct_b, B_ * 1024);
        gemm(ct_b, 1024, wihc_b, 1024, gD, 4096, b_d, B_, 4096, 4096, 1024, 0);
        gemm(e_bf + (size_t)t * B_ * EP_, EP_, wihe_b, EP_, gD, 4096, nullptr, B_, 4096, 4096, EP_, 1);
        gemm(hDb, 1024, whhd_b, 1024, gD, 4096, nullptr, B_, 4096, 4096, 1024, 1);
        k_lstm<<<dim3(g1k), dim3(256), 0, stream>>>(gD, nullptr, cD, hD, cD, hDb, nullptr, 1024);
        float* orow = out + (size_t)(t + 1) * B_ * V_;
        gemm(hDb, 1024, wgen_b, 1024, orow, V_, bgen, B_, V_, VP_, 1024, 0);
        k_logsoftmax<<<dim3(B_), dim3(256), 0, stream>>>(orow);
    }
}
